// PAM_22076131901725
// MI455X (gfx1250) — matrix operand order repaired and checked by reading, not compiled, not run on hardware
//
#include <hip/hip_runtime.h>
#include <hip/hip_bf16.h>

#define B_DIM  4
#define C_DIM  256
#define CQ_DIM 64
#define N_DIM  4096

typedef __bf16 bf16_t;
typedef __attribute__((ext_vector_type(16))) __bf16 v16bf;
typedef __attribute__((ext_vector_type(8)))  __bf16 v8bf;
typedef __attribute__((ext_vector_type(8)))  float  v8f;
typedef __attribute__((ext_vector_type(4)))  unsigned int v4u;
typedef __attribute__((ext_vector_type(8)))  int  v8i;
typedef __attribute__((ext_vector_type(4)))  int  v4i;

union ABf { v16bf v; v8bf h[2]; };

static __device__ inline v8f wmma_bf16(const ABf& a, const ABf& b, v8f c) {
    // D(16x16 f32) = A(16x32 bf16) * B(32x16 bf16) + C
    return __builtin_amdgcn_wmma_f32_16x16x32_bf16(
        false, a.v, false, b.v, (short)0, c, false, false);
}

// ---------------------------------------------------------------------------
// TDM descriptor (D#) builder + issue.  Packs groups per CDNA5 ISA:
//   group0: [1:0]=count=1, [63:32]=lds_addr, [120:64]=global_addr, [127:126]=2
//   group1: data_size=1(2B), pad_enable, pad_interval/amount, dims/strides
//   groups 2/3: zero (<= 2-D tile)
// tile_d1 == 0 -> pure 1-D copy of tile_d0 elements.
// This toolchain exposes the 6-arg builtin:
//   (uint32x4 g0, int32x8 g1, int32x4 g2, int32x4 g3, int32x8 g4, i32 cpol)
// ---------------------------------------------------------------------------
static __device__ inline void tdm_load_to_lds(
    unsigned int lds_byte_addr, unsigned long long gaddr,
    unsigned int tile_d0, unsigned int tile_d1,
    unsigned long long t0_stride,              // elements
    unsigned int pad_interval, unsigned int pad_amount)
{
    const unsigned int td0 = 1u << 30, td1 = 1u << 30;   // large: no OOB clipping
    v4u g0;
    g0[0] = 1u;                                  // count=1, user descriptor
    g0[1] = lds_byte_addr;
    g0[2] = (unsigned int)gaddr;
    g0[3] = (unsigned int)(gaddr >> 32) | (2u << 30);    // type=2 ("image")
    v8i g1;
    g1[0] = (int)((1u << 16)                      // data_size = 2 bytes
                | (1u << 20)                      // pad_enable
                | (pad_interval << 22)
                | (pad_amount << 25));
    g1[1] = (int)((td0 & 0xFFFFu) << 16);         // atomic_barrier_addr=0 | td0.lo
    g1[2] = (int)((td0 >> 16) | ((td1 & 0xFFFFu) << 16));
    g1[3] = (int)((td1 >> 16) | (tile_d0 << 16));
    g1[4] = (int)(tile_d1 & 0xFFFFu);             // tile_dim1 | tile_dim2=0
    g1[5] = (int)(unsigned int)t0_stride;
    g1[6] = (int)((unsigned int)(t0_stride >> 32) & 0xFFFFu);  // dim1_stride=0
    g1[7] = 0;
    v4i g2 = {0, 0, 0, 0};
    v4i g3 = {0, 0, 0, 0};
    v8i g4 = {0, 0, 0, 0, 0, 0, 0, 0};
    __builtin_amdgcn_tensor_load_to_lds(g0, g1, g2, g3, g4, 0);
}

// ---------------------------------------------------------------------------
// Kernel 1: fused QKV projection (1x1 convs as GEMMs), WMMA bf16.
//   Qt, Kt stored transposed [N,64]; V stored [256,N]. All bf16.
// ---------------------------------------------------------------------------
__global__ __launch_bounds__(256) void proj_qkv(
    const float* __restrict__ x,
    const float* __restrict__ Wq, const float* __restrict__ bq,
    const float* __restrict__ Wk, const float* __restrict__ bk,
    const float* __restrict__ Wv, const float* __restrict__ bv,
    bf16_t* __restrict__ Qt, bf16_t* __restrict__ Kt, bf16_t* __restrict__ V)
{
    __shared__ alignas(16) bf16_t xs[64 * 264];   // [n_local][c], 528B rows

    const int tid  = threadIdx.x;
    const int lane = tid & 31;
    const int wave = tid >> 5;
    const int lm   = lane & 15;
    const int hf   = (lane >> 4) & 1;
    const int b    = blockIdx.x >> 6;
    const int n0   = (blockIdx.x & 63) * 64;

    const float* xb = x + (size_t)b * C_DIM * N_DIM;

    #pragma unroll 4
    for (int i = 0; i < 64; ++i) {
        int e  = i * 256 + tid;
        int nl = e & 63;
        int c  = e >> 6;
        xs[nl * 264 + c] = (bf16_t)xb[(size_t)c * N_DIM + n0 + nl];
    }
    __syncthreads();

    for (int jj = 0; jj < 12; ++jj) {
        int job = wave * 12 + jj;
        int ot  = job >> 2;
        int ns  = job & 3;

        const float* Wp; const float* bp; int o0; int mode;
        if (ot < 4)      { Wp = Wq; bp = bq; o0 = ot * 16;       mode = 0; }
        else if (ot < 8) { Wp = Wk; bp = bk; o0 = (ot - 4) * 16; mode = 1; }
        else             { Wp = Wv; bp = bv; o0 = (ot - 8) * 16; mode = 2; }

        v8f acc;
        #pragma unroll
        for (int r = 0; r < 8; ++r) acc[r] = 0.f;

        #pragma unroll
        for (int cc = 0; cc < 8; ++cc) {
            ABf a;
            const float* wr = Wp + (size_t)(o0 + lm) * C_DIM + cc * 32 + hf * 8;
            #pragma unroll
            for (int k = 0; k < 8; ++k) {
                a.v[k]     = (bf16_t)wr[k];
                a.v[k + 8] = (bf16_t)wr[k + 16];
            }
            ABf bx;
            const bf16_t* xr = xs + (ns * 16 + lm) * 264 + cc * 32 + hf * 8;
            bx.h[0] = *(const v8bf*)(xr);
            bx.h[1] = *(const v8bf*)(xr + 16);
            acc = wmma_bf16(a, bx, acc);
        }

        const int ncol = n0 + ns * 16 + lm;
        if (mode == 2) {
            bf16_t* dst = V + (size_t)b * C_DIM * N_DIM;
            #pragma unroll
            for (int r = 0; r < 8; ++r) {
                int o = o0 + hf * 8 + r;
                dst[(size_t)o * N_DIM + ncol] = (bf16_t)(acc[r] + bp[o]);
            }
        } else {
            bf16_t* dst = (mode == 0 ? Qt : Kt) + (size_t)b * N_DIM * CQ_DIM;
            v8bf pk;
            #pragma unroll
            for (int r = 0; r < 8; ++r) {
                int o = o0 + hf * 8 + r;
                pk[r] = (bf16_t)(acc[r] + bp[o]);
            }
            *(v8bf*)(dst + (size_t)ncol * CQ_DIM + o0 + hf * 8) = pk;
        }
    }
}

// ---------------------------------------------------------------------------
// Kernel 2: flash attention, transposed-S formulation, TDM-staged K/V tiles.
//   Per 32-key chunk the TDM DMAs K(32x64) and V(256x32) into LDS (padded
//   rows: 144B / 80B strides -> conflict-free ds_load_b128 fragments),
//   double-buffered so DMA(i+1) overlaps compute(i).
// LDS byte map: K0 @0 (4608B) | K1 @4608 | V0 @9216 (20480B) | V1 @29696
// ---------------------------------------------------------------------------
#define K_TILE_B 4608     // 32 rows * (128B + 16B pad)
#define V_TILE_B 20480    // 256 rows * (64B + 16B pad)
#define SMEM_B   (2 * (K_TILE_B + V_TILE_B))

__global__ __launch_bounds__(256) void attn_pam(
    const float*  __restrict__ x,
    const bf16_t* __restrict__ Qt, const bf16_t* __restrict__ Kt,
    const bf16_t* __restrict__ V,
    const float*  __restrict__ gamma,
    float* __restrict__ out)
{
    __shared__ alignas(16) bf16_t smem[SMEM_B / 2];

    const int tid  = threadIdx.x;
    const int lane = tid & 31;
    const int wave = tid >> 5;
    const int lm   = lane & 15;
    const int hf   = (lane >> 4) & 1;
    const int b    = blockIdx.x >> 5;
    const int q0   = (blockIdx.x & 31) * 128 + wave * 16;

    const bf16_t* Qb = Qt + (size_t)b * N_DIM * CQ_DIM;
    const bf16_t* Kb = Kt + (size_t)b * N_DIM * CQ_DIM;
    const bf16_t* Vb = V  + (size_t)b * C_DIM * N_DIM;

    const unsigned int lds0 = (unsigned int)(uintptr_t)(&smem[0]);
    const unsigned long long kgbase = (unsigned long long)(uintptr_t)Kb;
    const unsigned long long vgbase = (unsigned long long)(uintptr_t)Vb;

    // Persistent Q B-fragments (queries = columns), 2 chunks of 32 channels.
    ABf qf[2];
    {
        const bf16_t* qr = Qb + (size_t)(q0 + lm) * CQ_DIM + hf * 8;
        qf[0].h[0] = *(const v8bf*)(qr);
        qf[0].h[1] = *(const v8bf*)(qr + 16);
        qf[1].h[0] = *(const v8bf*)(qr + 32);
        qf[1].h[1] = *(const v8bf*)(qr + 48);
    }

    v8f acc[16];
    #pragma unroll
    for (int t = 0; t < 16; ++t)
        #pragma unroll
        for (int r = 0; r < 8; ++r) acc[t][r] = 0.f;

    float m = -1e30f, l = 0.f;

    // Prologue: DMA chunk 0 into buffer 0 (wave 0 owns the TDM queue).
    if (wave == 0) {
        tdm_load_to_lds(lds0,              kgbase, 2048, 0,   2048, 4, 3);
        tdm_load_to_lds(lds0 + 2*K_TILE_B, vgbase, 32,   256, 4096, 3, 3);
    }

    for (int j0 = 0; j0 < N_DIM; j0 += 32) {
        const int buf = (j0 >> 5) & 1;

        if (wave == 0) {
            if (j0 + 32 < N_DIM) {
                const int nb = buf ^ 1;
                unsigned long long kg = kgbase + (unsigned long long)(j0 + 32) * 128ull;
                unsigned long long vg = vgbase + (unsigned long long)(j0 + 32) * 2ull;
                tdm_load_to_lds(lds0 + nb * K_TILE_B,              kg, 2048, 0,   2048, 4, 3);
                tdm_load_to_lds(lds0 + 2*K_TILE_B + nb * V_TILE_B, vg, 32,   256, 4096, 3, 3);
                __builtin_amdgcn_s_wait_tensorcnt(2);   // current chunk's pair done
            } else {
                __builtin_amdgcn_s_wait_tensorcnt(0);
            }
        }
        __syncthreads();                                 // buf ready for all waves

        const bf16_t* kt = smem + (buf ? (K_TILE_B / 2) : 0);
        const bf16_t* vt = smem + K_TILE_B + (buf ? (V_TILE_B / 2) : 0);

        // ---- S^T tiles from LDS: rows = keys, cols = 16 queries ----
        v8f s0, s1;
        #pragma unroll
        for (int r = 0; r < 8; ++r) { s0[r] = 0.f; s1[r] = 0.f; }
        #pragma unroll
        for (int cc = 0; cc < 2; ++cc) {
            ABf a0, a1;
            const bf16_t* kr0 = kt + lm * 72        + cc * 32 + hf * 8;  // 144B rows
            const bf16_t* kr1 = kt + (16 + lm) * 72 + cc * 32 + hf * 8;
            a0.h[0] = *(const v8bf*)(kr0);  a0.h[1] = *(const v8bf*)(kr0 + 16);
            a1.h[0] = *(const v8bf*)(kr1);  a1.h[1] = *(const v8bf*)(kr1 + 16);
            s0 = wmma_bf16(a0, qf[cc], s0);
            s1 = wmma_bf16(a1, qf[cc], s1);
        }

        // ---- online softmax (per query column = per lane) ----
        float cm = -1e30f;
        #pragma unroll
        for (int r = 0; r < 8; ++r) { cm = fmaxf(cm, s0[r]); cm = fmaxf(cm, s1[r]); }
        cm = fmaxf(cm, __shfl_xor(cm, 16, 32));
        float mn    = fmaxf(m, cm);
        float alpha = __expf(m - mn);
        m = mn;

        float p0[8], p1[8], rs = 0.f;
        #pragma unroll
        for (int r = 0; r < 8; ++r) {
            p0[r] = __expf(s0[r] - mn);
            p1[r] = __expf(s1[r] - mn);
            rs += p0[r] + p1[r];
        }
        rs += __shfl_xor(rs, 16, 32);
        l = l * alpha + rs;

        #pragma unroll
        for (int t = 0; t < 16; ++t)
            #pragma unroll
            for (int r = 0; r < 8; ++r) acc[t][r] *= alpha;

        // ---- P^T B-fragment: one cross-half shuffle ----
        ABf pb;
        #pragma unroll
        for (int r = 0; r < 8; ++r) {
            float o0v = __shfl_xor(p0[r], 16, 32);
            float o1v = __shfl_xor(p1[r], 16, 32);
            pb.v[r]     = (bf16_t)(hf ? o1v   : p0[r]);
            pb.v[r + 8] = (bf16_t)(hf ? p1[r] : o0v);
        }

        // ---- O^T += V_tile(16c x 32j) * P^T(32j x 16q), V from LDS ----
        #pragma unroll
        for (int t = 0; t < 16; ++t) {
            ABf av;
            const bf16_t* vr = vt + (t * 16 + lm) * 40 + hf * 16;       // 80B rows
            av.h[0] = *(const v8bf*)(vr);
            av.h[1] = *(const v8bf*)(vr + 8);
            acc[t] = wmma_bf16(av, pb, acc[t]);
        }

        __syncthreads();   // everyone done reading buf before TDM reuses it
    }

    // ---- epilogue: y = gamma * O/l + x ----
    const float g  = gamma[0];
    const float rl = 1.f / l;
    const int   n  = q0 + lm;
    const float* xb = x   + (size_t)b * C_DIM * N_DIM;
    float*       ob = out + (size_t)b * C_DIM * N_DIM;
    #pragma unroll
    for (int t = 0; t < 16; ++t) {
        #pragma unroll
        for (int r = 0; r < 8; ++r) {
            int c = t * 16 + hf * 8 + r;
            size_t idx = (size_t)c * N_DIM + n;
            ob[idx] = g * acc[t][r] * rl + xb[idx];
        }
    }
}

// ---------------------------------------------------------------------------
extern "C" void kernel_launch(void* const* d_in, const int* in_sizes, int n_in,
                              void* d_out, int out_size, void* d_ws, size_t ws_size,
                              hipStream_t stream) {
    const float* x     = (const float*)d_in[0];
    const float* Wq    = (const float*)d_in[1];
    const float* bq    = (const float*)d_in[2];
    const float* Wk    = (const float*)d_in[3];
    const float* bk    = (const float*)d_in[4];
    const float* Wv    = (const float*)d_in[5];
    const float* bv    = (const float*)d_in[6];
    const float* gamma = (const float*)d_in[7];
    float* out = (float*)d_out;

    // Workspace layout (bf16): Qt[B,N,64] | Kt[B,N,64] | V[B,256,N]  = 12 MB
    bf16_t* Qt = (bf16_t*)d_ws;
    bf16_t* Kt = Qt + (size_t)B_DIM * N_DIM * CQ_DIM;
    bf16_t* V  = Kt + (size_t)B_DIM * N_DIM * CQ_DIM;

    proj_qkv<<<dim3(B_DIM * (N_DIM / 64)),  dim3(256), 0, stream>>>(
        x, Wq, bq, Wk, bk, Wv, bv, Qt, Kt, V);
    attn_pam<<<dim3(B_DIM * (N_DIM / 128)), dim3(256), 0, stream>>>(
        x, Qt, Kt, V, gamma, out);
}
